// UnrolledMSENet_38689065402872
// MI455X (gfx1250) — compile-verified, matrix operand order checked by reading, not run on hardware
//
#include <hip/hip_runtime.h>

// ---------------------------------------------------------------------------
// CDNA5 / gfx1250 bf16-WMMA implementation of UnrolledMSENet
//   wave32, v_wmma_f32_16x16x32_bf16, fp32 accumulate, fp32 state.
// ---------------------------------------------------------------------------

typedef __bf16 bf16_t;
typedef bf16_t v16bf __attribute__((ext_vector_type(16)));
typedef bf16_t v8bf  __attribute__((ext_vector_type(8)));
typedef float  v8f   __attribute__((ext_vector_type(8)));

#define B_ROWS 4096
#define LFIX   8192
#define NFRM   2112
#define MID    264
#define MIDP   288   // MID padded to multiple of 32 (K) and 16 (N)
#define PDIM   88
#define PP     96    // P padded to multiple of 32

// Epilogue modes for the generic GEMM
#define MODE_XI   0  // xi = 1.6*(acc - feat) + 0.2*xi_old           (fp32 + bf16 out)
#define MODE_M    1  // m  = relu(cA*acc + cB*m_old)                 (fp32 + bf16 out)
#define MODE_RELU 2  // h  = relu(acc + bias[n])                     (bf16 out)
#define MODE_SIG  3  // o  = sigmoid(acc + bias[n])                  (fp32 out)

// Load one 16-bit WMMA operand fragment (A or B, identical NT layout):
// lane holds row (lane&15); K elements: [kh+0..7] and [kh+16..23], kh=(lane>>4)*8.
// p must already point at rowBase + k0 + kh.
static __device__ __forceinline__ v16bf load_frag16(const bf16_t* p) {
  union { v16bf v; v8bf h[2]; } u;
  u.h[0] = *(const v8bf*)(p);        // K = k0+kh .. +7
  u.h[1] = *(const v8bf*)(p + 16);   // K = k0+kh+16 .. +23
  return u.v;
}

static __device__ __forceinline__ v8f zero_v8f() {
  v8f z;
#pragma unroll
  for (int t = 0; t < 8; ++t) z[t] = 0.0f;
  return z;
}

// ---------------------------------------------------------------------------
// scale2[row] = (LFIX / (count_nonzero(row)+1))^2
// ---------------------------------------------------------------------------
__global__ void k_scale2(const float* __restrict__ x, float* __restrict__ scale2, int len) {
  __shared__ int red[256];
  const int row = blockIdx.x;
  const float* p = x + (size_t)row * len;
  int cnt = 0;
  for (int i = threadIdx.x; i < len; i += blockDim.x) cnt += (p[i] != 0.0f);
  red[threadIdx.x] = cnt;
  __syncthreads();
  for (int s = 128; s > 0; s >>= 1) {
    if (threadIdx.x < s) red[threadIdx.x] += red[threadIdx.x + s];
    __syncthreads();
  }
  if (threadIdx.x == 0) {
    float sc = (float)len / (float)(red[0] + 1);
    scale2[row] = sc * sc;
  }
}

// ---------------------------------------------------------------------------
// fp32 -> bf16 conversion, 8 elements per thread (n must be a multiple of 8)
// ---------------------------------------------------------------------------
__global__ void k_cvt8(const float* __restrict__ in, bf16_t* __restrict__ out, long n) {
  long i = ((long)blockIdx.x * blockDim.x + threadIdx.x) * 8;
  const long stride = (long)gridDim.x * blockDim.x * 8;
  for (; i < n; i += stride) {
    float4 a = *(const float4*)(in + i);
    float4 b = *(const float4*)(in + i + 4);
    v8bf o;
    o[0] = (bf16_t)a.x; o[1] = (bf16_t)a.y; o[2] = (bf16_t)a.z; o[3] = (bf16_t)a.w;
    o[4] = (bf16_t)b.x; o[5] = (bf16_t)b.y; o[6] = (bf16_t)b.z; o[7] = (bf16_t)b.w;
    *(v8bf*)(out + i) = o;
  }
}

// dst[dr x dc] (bf16, zero-padded) = src[sr x sc] (fp32)
__global__ void k_pad2d(const float* __restrict__ src, int sr, int sc,
                        bf16_t* __restrict__ dst, int dr, int dc) {
  int idx = blockIdx.x * blockDim.x + threadIdx.x;
  int total = dr * dc;
  if (idx >= total) return;
  int r = idx / dc, c = idx % dc;
  dst[idx] = (r < sr && c < sc) ? (bf16_t)src[r * sc + c] : (bf16_t)0.0f;
}

// dst[dr x dc] = zero-padded transpose of src[sr x sc]: dst[r][c] = src[c][r]
__global__ void k_pad2dT(const float* __restrict__ src, int sr, int sc,
                         bf16_t* __restrict__ dst, int dr, int dc) {
  int idx = blockIdx.x * blockDim.x + threadIdx.x;
  int total = dr * dc;
  if (idx >= total) return;
  int r = idx / dc, c = idx % dc;
  dst[idx] = (r < sc && c < sr) ? (bf16_t)src[c * sc + r] : (bf16_t)0.0f;
}

__global__ void k_pad1d(const float* __restrict__ src, int n, float* __restrict__ dst, int np) {
  int i = blockIdx.x * blockDim.x + threadIdx.x;
  if (i < np) dst[i] = (i < n) ? src[i] : 0.0f;
}

// ---------------------------------------------------------------------------
// k_feat: fused dual GEMM  (feat = ((A@Wc^T)^2 + (A@Ws^T)^2) * scale2[row])
//   A  [4096 x 8192] bf16, Wc/Ws [2112 x 8192] bf16 (NT: K contiguous both)
//   outputs: feat fp32, xi = -feat (fp32 + bf16)
// Workgroup = 8 waves (4M x 2N), wave tile 64x32, WG tile 256x64.
// ---------------------------------------------------------------------------
__global__ __launch_bounds__(256)
void k_feat(const bf16_t* __restrict__ A,
            const bf16_t* __restrict__ Wc,
            const bf16_t* __restrict__ Ws,
            const float* __restrict__ scale2,
            float* __restrict__ feat,
            float* xiF, bf16_t* xiB) {
  const int lane = threadIdx.x & 31;
  const int wave = threadIdx.x >> 5;
  const int wm = wave & 3;   // 0..3
  const int wn = wave >> 2;  // 0..1
  const int m0 = blockIdx.x * 256 + wm * 64;
  const int n0 = blockIdx.y * 64 + wn * 32;
  const int r  = lane & 15;
  const int kh = (lane >> 4) * 8;

  v8f accC[4][2], accS[4][2];
#pragma unroll
  for (int i = 0; i < 4; ++i)
#pragma unroll
    for (int j = 0; j < 2; ++j) { accC[i][j] = zero_v8f(); accS[i][j] = zero_v8f(); }

  const bf16_t* Arow[4];
#pragma unroll
  for (int i = 0; i < 4; ++i) Arow[i] = A + (size_t)(m0 + i * 16 + r) * LFIX + kh;
  const bf16_t* Wcr[2];
  const bf16_t* Wsr[2];
#pragma unroll
  for (int j = 0; j < 2; ++j) {
    const int col = n0 + j * 16 + r;   // always < 2112 (33 exact tiles)
    Wcr[j] = Wc + (size_t)col * LFIX + kh;
    Wsr[j] = Ws + (size_t)col * LFIX + kh;
  }

  for (int ks = 0; ks < LFIX / 32; ++ks) {
    const int k0 = ks * 32;
    v16bf bc0 = load_frag16(Wcr[0] + k0);
    v16bf bc1 = load_frag16(Wcr[1] + k0);
    v16bf bs0 = load_frag16(Wsr[0] + k0);
    v16bf bs1 = load_frag16(Wsr[1] + k0);
#pragma unroll
    for (int i = 0; i < 4; ++i) {
      v16bf a = load_frag16(Arow[i] + k0);
      accC[i][0] = __builtin_amdgcn_wmma_f32_16x16x32_bf16(false, a, false, bc0, (short)0, accC[i][0], false, false);
      accC[i][1] = __builtin_amdgcn_wmma_f32_16x16x32_bf16(false, a, false, bc1, (short)0, accC[i][1], false, false);
      accS[i][0] = __builtin_amdgcn_wmma_f32_16x16x32_bf16(false, a, false, bs0, (short)0, accS[i][0], false, false);
      accS[i][1] = __builtin_amdgcn_wmma_f32_16x16x32_bf16(false, a, false, bs1, (short)0, accS[i][1], false, false);
    }
  }

  // C/D layout: VGPR v, lanes 0-15 -> M=v, lanes 16-31 -> M=v+8; N = lane&15
  const int rowoff = (lane >> 4) * 8;
  const int ncol = lane & 15;
#pragma unroll
  for (int i = 0; i < 4; ++i)
#pragma unroll
    for (int j = 0; j < 2; ++j) {
      const int col = n0 + j * 16 + ncol;
#pragma unroll
      for (int v = 0; v < 8; ++v) {
        const int row = m0 + i * 16 + rowoff + v;
        const float c = accC[i][j][v];
        const float s = accS[i][j][v];
        const float f = (c * c + s * s) * scale2[row];
        const size_t idx = (size_t)row * NFRM + col;
        feat[idx] = f;
        xiF[idx] = -f;
        xiB[idx] = (bf16_t)(-f);
      }
    }
}

// ---------------------------------------------------------------------------
// Generic NT bf16 GEMM with fused epilogues.
//   A [4096 x lda], W [Npad x ldw] (row-major, K contiguous), K = Ksteps*32.
//   MODE_XI:   aux0=feat (ld=ldF), aux1=xi_old;  out: xi fp32 + bf16
//   MODE_M:    aux1=m_old (ld=ldF);              out: m fp32 + bf16
//   MODE_RELU: aux0=bias[n];                     out: bf16
//   MODE_SIG:  aux0=bias[n];                     out: fp32 (cols < NoutF)
// ---------------------------------------------------------------------------
template <int MODE>
__global__ __launch_bounds__(256)
void k_gemm(const bf16_t* __restrict__ A, int lda,
            const bf16_t* __restrict__ W, int ldw,
            int Npad, int Ksteps,
            const float* __restrict__ aux0,
            const float* aux1,
            float* outF, int ldF, int NoutF,
            bf16_t* outB, int ldB,
            float cA, float cB) {
  const int lane = threadIdx.x & 31;
  const int wave = threadIdx.x >> 5;
  const int wm = wave & 3;
  const int wn = wave >> 2;
  const int m0 = blockIdx.x * 256 + wm * 64;
  const int n0 = blockIdx.y * 64 + wn * 32;
  const int r  = lane & 15;
  const int kh = (lane >> 4) * 8;

  v8f acc[4][2];
#pragma unroll
  for (int i = 0; i < 4; ++i)
#pragma unroll
    for (int j = 0; j < 2; ++j) acc[i][j] = zero_v8f();

  const bf16_t* Arow[4];
#pragma unroll
  for (int i = 0; i < 4; ++i) Arow[i] = A + (size_t)(m0 + i * 16 + r) * lda + kh;
  const bf16_t* Wrow[2];
#pragma unroll
  for (int j = 0; j < 2; ++j) {
    int col = n0 + j * 16 + r;
    if (col >= Npad) col = 0;  // clamp: loads stay in-bounds, stores guarded below
    Wrow[j] = W + (size_t)col * ldw + kh;
  }

  for (int ks = 0; ks < Ksteps; ++ks) {
    const int k0 = ks * 32;
    v16bf b0 = load_frag16(Wrow[0] + k0);
    v16bf b1 = load_frag16(Wrow[1] + k0);
#pragma unroll
    for (int i = 0; i < 4; ++i) {
      v16bf a = load_frag16(Arow[i] + k0);
      acc[i][0] = __builtin_amdgcn_wmma_f32_16x16x32_bf16(false, a, false, b0, (short)0, acc[i][0], false, false);
      acc[i][1] = __builtin_amdgcn_wmma_f32_16x16x32_bf16(false, a, false, b1, (short)0, acc[i][1], false, false);
    }
  }

  const int rowoff = (lane >> 4) * 8;
  const int ncol = lane & 15;
#pragma unroll
  for (int i = 0; i < 4; ++i)
#pragma unroll
    for (int j = 0; j < 2; ++j) {
      const int col = n0 + j * 16 + ncol;
      if (col >= Npad) continue;
#pragma unroll
      for (int v = 0; v < 8; ++v) {
        const int row = m0 + i * 16 + rowoff + v;
        const float x = acc[i][j][v];
        float val;
        if constexpr (MODE == MODE_XI) {
          const size_t idx = (size_t)row * ldF + col;
          val = 1.6f * (x - aux0[idx]) + 0.2f * aux1[idx];
        } else if constexpr (MODE == MODE_M) {
          const size_t idx = (size_t)row * ldF + col;
          const float t = cA * x + cB * aux1[idx];
          val = t > 0.0f ? t : 0.0f;
        } else if constexpr (MODE == MODE_RELU) {
          const float t = x + aux0[col];
          val = t > 0.0f ? t : 0.0f;
        } else {  // MODE_SIG
          const float t = x + aux0[col];
          val = 1.0f / (1.0f + __expf(-t));
        }
        if (outF && col < NoutF) outF[(size_t)row * ldF + col] = val;
        if (outB) outB[(size_t)row * ldB + col] = (bf16_t)val;
      }
    }
}

// ---------------------------------------------------------------------------
// Host-side orchestration
// ---------------------------------------------------------------------------
extern "C" void kernel_launch(void* const* d_in, const int* in_sizes, int n_in,
                              void* d_out, int out_size, void* d_ws, size_t ws_size,
                              hipStream_t stream) {
  const float* in_this = (const float*)d_in[0];
  // d_in[1] = input_prev : unused by the reference
  const float* W_cos = (const float*)d_in[2];
  const float* W_sin = (const float*)d_in[3];
  const float* tmpl  = (const float*)d_in[4];
  const float* W1    = (const float*)d_in[5];
  const float* b1    = (const float*)d_in[6];
  const float* W2    = (const float*)d_in[7];
  const float* b2    = (const float*)d_in[8];
  const float* Wout  = (const float*)d_in[9];
  const float* bout  = (const float*)d_in[10];
  float* out = (float*)d_out;

  // Workspace carve-out (~235 MB), 256B aligned.
  char* base = (char*)d_ws;
  size_t off = 0;
  auto take = [&](size_t bytes) -> char* {
    off = (off + 255) & ~(size_t)255;
    char* p = base + off;
    off += bytes;
    return p;
  };
  float*  scale2 = (float*)take((size_t)B_ROWS * 4);
  bf16_t* Abf    = (bf16_t*)take((size_t)B_ROWS * LFIX * 2);
  bf16_t* Wcb    = (bf16_t*)take((size_t)NFRM * LFIX * 2);
  bf16_t* Wsb    = (bf16_t*)take((size_t)NFRM * LFIX * 2);
  bf16_t* Tb     = (bf16_t*)take((size_t)NFRM * MIDP * 2);   // template, K-pad (for m@T^T)
  bf16_t* TTb    = (bf16_t*)take((size_t)MIDP * NFRM * 2);   // template^T, N-pad (for xi@T)
  bf16_t* W1b    = (bf16_t*)take((size_t)MIDP * MIDP * 2);
  bf16_t* W2b    = (bf16_t*)take((size_t)PP * MIDP * 2);
  bf16_t* Wob    = (bf16_t*)take((size_t)PP * PP * 2);
  float*  b1p    = (float*)take((size_t)MIDP * 4);
  float*  b2p    = (float*)take((size_t)PP * 4);
  float*  bop    = (float*)take((size_t)PP * 4);
  float*  featF  = (float*)take((size_t)B_ROWS * NFRM * 4);
  float*  xiF    = (float*)take((size_t)B_ROWS * NFRM * 4);
  bf16_t* xiB    = (bf16_t*)take((size_t)B_ROWS * NFRM * 2);
  float*  mF     = (float*)take((size_t)B_ROWS * MIDP * 4);
  bf16_t* mB     = (bf16_t*)take((size_t)B_ROWS * MIDP * 2);
  bf16_t* h1B    = (bf16_t*)take((size_t)B_ROWS * MIDP * 2);
  bf16_t* h2B    = (bf16_t*)take((size_t)B_ROWS * PP * 2);
  (void)ws_size; (void)in_sizes; (void)n_in; (void)out_size;

  // 1. Row scales
  k_scale2<<<B_ROWS, 256, 0, stream>>>(in_this, scale2, LFIX);

  // 2. bf16 conversions / padded weights
  k_cvt8<<<8192, 256, 0, stream>>>(in_this, Abf, (long)B_ROWS * LFIX);
  k_cvt8<<<8192, 256, 0, stream>>>(W_cos, Wcb, (long)NFRM * LFIX);
  k_cvt8<<<8192, 256, 0, stream>>>(W_sin, Wsb, (long)NFRM * LFIX);
  k_pad2d<<<(NFRM * MIDP + 255) / 256, 256, 0, stream>>>(tmpl, NFRM, MID, Tb, NFRM, MIDP);
  k_pad2dT<<<(MIDP * NFRM + 255) / 256, 256, 0, stream>>>(tmpl, NFRM, MID, TTb, MIDP, NFRM);
  k_pad2d<<<(MIDP * MIDP + 255) / 256, 256, 0, stream>>>(W1, MID, MID, W1b, MIDP, MIDP);
  k_pad2d<<<(PP * MIDP + 255) / 256, 256, 0, stream>>>(W2, PDIM, MID, W2b, PP, MIDP);
  k_pad2d<<<(PP * PP + 255) / 256, 256, 0, stream>>>(Wout, PDIM, PDIM, Wob, PP, PP);
  k_pad1d<<<(MIDP + 255) / 256, 256, 0, stream>>>(b1, MID, b1p, MIDP);
  k_pad1d<<<1, 256, 0, stream>>>(b2, PDIM, b2p, PP);
  k_pad1d<<<1, 256, 0, stream>>>(bout, PDIM, bop, PP);

  // 3. Fused cos+sin GEMM -> feat, xi0 = -feat
  k_feat<<<dim3(B_ROWS / 256, NFRM / 64), 256, 0, stream>>>(Abf, Wcb, Wsb, scale2,
                                                            featF, xiF, xiB);

  const dim3 gM(B_ROWS / 256, (MIDP + 63) / 64);   // N = 288 -> 5 tiles (guarded)
  const dim3 gX(B_ROWS / 256, NFRM / 64);          // N = 2112 -> 33 tiles
  const dim3 gP(B_ROWS / 256, (PP + 63) / 64);     // N = 96 -> 2 tiles (guarded)

  // 4. m = relu(-10 * (xi0 @ template))          [K=2112, N=288]
  k_gemm<MODE_M><<<gM, 256, 0, stream>>>(xiB, NFRM, TTb, NFRM, MIDP, NFRM / 32,
                                         nullptr, mF, mF, MIDP, MIDP, mB, MIDP,
                                         -10.0f, 0.0f);
  // 5. Four LISTA iterations
  for (int it = 0; it < 4; ++it) {
    // xi = 1.6*((m @ template^T) - feat) + 0.2*xi   [K=288, N=2112]
    k_gemm<MODE_XI><<<gX, 256, 0, stream>>>(mB, MIDP, Tb, MIDP, NFRM, MIDP / 32,
                                            featF, xiF, xiF, NFRM, NFRM, xiB, NFRM,
                                            0.0f, 0.0f);
    // m = relu(-8*(xi @ template) + 0.2*m)          [K=2112, N=288]
    k_gemm<MODE_M><<<gM, 256, 0, stream>>>(xiB, NFRM, TTb, NFRM, MIDP, NFRM / 32,
                                           nullptr, mF, mF, MIDP, MIDP, mB, MIDP,
                                           -8.0f, 0.2f);
  }

  // 6. MLP head
  k_gemm<MODE_RELU><<<gM, 256, 0, stream>>>(mB, MIDP, W1b, MIDP, MIDP, MIDP / 32,
                                            b1p, nullptr, nullptr, 0, 0, h1B, MIDP,
                                            0.0f, 0.0f);
  k_gemm<MODE_RELU><<<gP, 256, 0, stream>>>(h1B, MIDP, W2b, MIDP, PP, MIDP / 32,
                                            b2p, nullptr, nullptr, 0, 0, h2B, PP,
                                            0.0f, 0.0f);
  k_gemm<MODE_SIG><<<gP, 256, 0, stream>>>(h2B, PP, Wob, PP, PP, PP / 32,
                                           bop, nullptr, out, PDIM, PDIM, nullptr, 0,
                                           0.0f, 0.0f);
}
